// QLSTM_65481071404693
// MI455X (gfx1250) — matrix operand order repaired and checked by reading, not compiled, not run on hardware
//
#include <hip/hip_runtime.h>
#include <math.h>

// CDNA5 / gfx1250, wave32. WMMA bf16 path.
typedef __attribute__((ext_vector_type(16))) __bf16 v16bf;
typedef __attribute__((ext_vector_type(8)))  float  v8f;

constexpr int T_ = 256, B_ = 128, D_ = 1024, H_ = 1024;
constexpr int KTOT   = D_ + H_;     // 2048
constexpr int KTILES = KTOT / 32;   // 64 K-chunks of 32
constexpr int NTILES = H_ / 16;     // 64 column tiles of 16
// Packed W: [3 gates][NTILES][KTILES][32 lanes][16 bf16]  (f,i,g only; o-gate is dead)
constexpr size_t PACKED_ELEMS = (size_t)3 * NTILES * KTILES * 32 * 16; // 6,291,456
constexpr size_t PACKED_BYTES = PACKED_ELEMS * 2;                      // 12,582,912

// ---------------------------------------------------------------------------
// One-time repack of W (f32 row-major [2048,1024] per gate) into bf16
// fragment-major layout matching the WMMA B operand (32x16, K x N):
//   lane half (lane>>4) selects K runs 8*half..8*half+7 and 16+8*half..16+8*half+7
//   element j within the lane's v16bf is K = 32*kt + 8*half + (j&7) + 16*(j>>3), N = nt*16 + (lane&15)
// ---------------------------------------------------------------------------
__global__ void pack_w(const float* __restrict__ Wf, const float* __restrict__ Wi,
                       const float* __restrict__ Wg, __bf16* __restrict__ packed) {
  size_t idx = (size_t)blockIdx.x * blockDim.x + threadIdx.x;
  if (idx >= PACKED_ELEMS) return;
  int j    = (int)(idx        & 15);
  int lane = (int)((idx >> 4) & 31);
  int kt   = (int)((idx >> 9) & 63);
  int nt   = (int)((idx >> 15) & 63);
  int gi   = (int)(idx >> 21);
  int half = lane >> 4;
  int n = nt * 16 + (lane & 15);
  int k = kt * 32 + 8 * half + (j & 7) + ((j >> 3) << 4);
  const float* W = (gi == 0) ? Wf : (gi == 1) ? Wi : Wg;
  packed[idx] = (__bf16)W[(size_t)k * H_ + n];
}

__global__ void zero_h(float* __restrict__ h) {
  int i = blockIdx.x * blockDim.x + threadIdx.x;
  h[i] = 0.0f;
}

// ---------------------------------------------------------------------------
// One recurrence step. Grid = NTILES (64) workgroups of 256 threads (8 waves).
// Wave w: rows 16w..16w+15; block: one 16-wide column strip of H.
// Three accumulators (f/i/g gates) share one A fragment per K-chunk.
// A fragment built per ISA 16-bit A layout: lane row = lane&15,
// K runs [32*kt + 8*half, +8) and [32*kt + 16 + 8*half, +8)  (f32->bf16 inline).
// ---------------------------------------------------------------------------
__global__ __launch_bounds__(256)
void lstm_step(const float* __restrict__ x,      // [T,B,D] f32
               const __bf16* __restrict__ pw,    // packed W (f,i,g)
               const float* __restrict__ bfv, const float* __restrict__ biv,
               const float* __restrict__ bgv,
               const float* __restrict__ hprev,  // [B,H] f32 (== c)
               float* __restrict__ hnext,        // [B,H] f32
               float* __restrict__ out,          // [T*B*H + 2*B*H] f32
               int t, int last) {
  const int tid  = threadIdx.x;
  const int lane = tid & 31;
  const int wave = tid >> 5;
  const int half = lane >> 4;
  const int l15  = lane & 15;
  const int row0 = wave * 16;
  const int col0 = blockIdx.x * 16;

  v8f accF = {}, accI = {}, accG = {};

  const float* xrow = x + ((size_t)t * B_ + (row0 + l15)) * D_;
  const float* hrow = hprev + (size_t)(row0 + l15) * H_;

  const v16bf* pwv = (const v16bf*)pw;  // one v16bf = one lane slice of a fragment
  const size_t fragStride = 32;         // v16bf units per fragment
  const size_t baseF = (((size_t)0 * NTILES + blockIdx.x) * KTILES) * fragStride + lane;
  const size_t baseI = (((size_t)1 * NTILES + blockIdx.x) * KTILES) * fragStride + lane;
  const size_t baseG = (((size_t)2 * NTILES + blockIdx.x) * KTILES) * fragStride + lane;

  for (int kt = 0; kt < KTILES; ++kt) {
    const int kc = kt * 32;
    const int k1 = kc + 8 * half;        // covers elements j=0..7
    const int k2 = k1 + 16;              // covers elements j=8..15
    // Branchless x/h split (boundary D_ is a multiple of 32; runs never straddle).
    const float* p1 = (k1 < D_) ? (xrow + k1) : (hrow + (k1 - D_));
    const float* p2 = (k2 < D_) ? (xrow + k2) : (hrow + (k2 - D_));
    float4 a0 = *(const float4*)(p1);
    float4 a1 = *(const float4*)(p1 + 4);
    float4 a2 = *(const float4*)(p2);
    float4 a3 = *(const float4*)(p2 + 4);

    v16bf a;
    a[0]  = (__bf16)a0.x; a[1]  = (__bf16)a0.y; a[2]  = (__bf16)a0.z; a[3]  = (__bf16)a0.w;
    a[4]  = (__bf16)a1.x; a[5]  = (__bf16)a1.y; a[6]  = (__bf16)a1.z; a[7]  = (__bf16)a1.w;
    a[8]  = (__bf16)a2.x; a[9]  = (__bf16)a2.y; a[10] = (__bf16)a2.z; a[11] = (__bf16)a2.w;
    a[12] = (__bf16)a3.x; a[13] = (__bf16)a3.y; a[14] = (__bf16)a3.z; a[15] = (__bf16)a3.w;

    v16bf bF = pwv[baseF + (size_t)kt * fragStride];
    v16bf bI = pwv[baseI + (size_t)kt * fragStride];
    v16bf bG = pwv[baseG + (size_t)kt * fragStride];

    if (kt + 1 < KTILES) {  // warm next fragments (L2 hits -> WGP$)
      __builtin_prefetch(&pwv[baseF + (size_t)(kt + 1) * fragStride], 0, 3);
      __builtin_prefetch(&pwv[baseI + (size_t)(kt + 1) * fragStride], 0, 3);
      __builtin_prefetch(&pwv[baseG + (size_t)(kt + 1) * fragStride], 0, 3);
    }

    accF = __builtin_amdgcn_wmma_f32_16x16x32_bf16(false, a, false, bF, (short)0, accF, false, false);
    accI = __builtin_amdgcn_wmma_f32_16x16x32_bf16(false, a, false, bI, (short)0, accI, false, false);
    accG = __builtin_amdgcn_wmma_f32_16x16x32_bf16(false, a, false, bG, (short)0, accG, false, false);
  }

  // Epilogue. C/D layout: VGPR r, lanes 0-15 -> M=r, lanes 16-31 -> M=r+8; N=lane&15.
  const int n = col0 + l15;
  const float bF = bfv[n], bI = biv[n], bG = bgv[n];
#pragma unroll
  for (int r = 0; r < 8; ++r) {
    const int m = row0 + r + 8 * half;
    const float zf = accF[r] + bF;
    const float zi = accI[r] + bI;
    const float zg = accG[r] + bG;
    const float fg = 1.0f / (1.0f + __expf(-zf));
    const float ig = 1.0f / (1.0f + __expf(-zi));
    const float gg = tanhf(zg);
    const float prev = hprev[(size_t)m * H_ + n];   // c == previous h
    const float nv = fg * prev + ig * gg;
    out[(size_t)t * B_ * H_ + (size_t)m * H_ + n] = nv;
    hnext[(size_t)m * H_ + n] = nv;
    if (last) {  // tuple tail: final h then final c (identical)
      out[(size_t)T_ * B_ * H_ + (size_t)m * H_ + n] = nv;
      out[(size_t)T_ * B_ * H_ + (size_t)B_ * H_ + (size_t)m * H_ + n] = nv;
    }
  }
}

extern "C" void kernel_launch(void* const* d_in, const int* in_sizes, int n_in,
                              void* d_out, int out_size, void* d_ws, size_t ws_size,
                              hipStream_t stream) {
  // setup_inputs order: inputs, W_f, b_f, W_i, b_i, W_g, b_g, W_o, b_o
  const float* x  = (const float*)d_in[0];
  const float* Wf = (const float*)d_in[1];
  const float* bf = (const float*)d_in[2];
  const float* Wi = (const float*)d_in[3];
  const float* bi = (const float*)d_in[4];
  const float* Wg = (const float*)d_in[5];
  const float* bg = (const float*)d_in[6];
  // d_in[7]=W_o, d_in[8]=b_o: dead in the reference, never touched.
  float* out = (float*)d_out;

  char* ws = (char*)d_ws;
  __bf16* packed = (__bf16*)ws;                                  // 12.6 MB
  float*  h0     = (float*)(ws + PACKED_BYTES);                  // 512 KB
  float*  h1     = (float*)(ws + PACKED_BYTES + (size_t)B_ * H_ * 4);

  pack_w<<<(int)(PACKED_ELEMS / 256), 256, 0, stream>>>(Wf, Wi, Wg, packed);
  zero_h<<<(B_ * H_) / 256, 256, 0, stream>>>(h0);

  for (int t = 0; t < T_; ++t) {
    const float* hp = (t & 1) ? h1 : h0;
    float*       hn = (t & 1) ? h0 : h1;
    lstm_step<<<NTILES, 256, 0, stream>>>(x, packed, bf, bi, bg, hp, hn, out,
                                          t, (t == T_ - 1) ? 1 : 0);
  }
}
